// SimVQ_48378511622626
// MI455X (gfx1250) — matrix operand order repaired and checked by reading, not compiled, not run on hardware
//
#include <hip/hip_runtime.h>
#include <hip/hip_bf16.h>

typedef __attribute__((ext_vector_type(16))) __bf16 v16bf;
typedef __attribute__((ext_vector_type(8)))  __bf16 v8bf;
typedef __attribute__((ext_vector_type(8)))  float  v8f;
typedef unsigned int u32x4 __attribute__((ext_vector_type(4)));
typedef int          i32x8 __attribute__((ext_vector_type(8)));
typedef int          i32x4 __attribute__((ext_vector_type(4)));

#define Bn 8
#define Dd 512
#define Nn 2048
#define Cc 8192
#define Mrows (Bn*Nn)        /* 16384 rows of xt */
#define QSIZE (Bn*Dd*Nn)     /* 8388608 floats of quantized output */

// ---------------------------------------------------------------------------
// f32 -> bf16 elementwise convert
// ---------------------------------------------------------------------------
__global__ void f32_to_bf16_kernel(const float* __restrict__ src,
                                   __bf16* __restrict__ dst, int n) {
  int i = blockIdx.x * blockDim.x + threadIdx.x;
  if (i < n) dst[i] = (__bf16)src[i];
}

// ---------------------------------------------------------------------------
// x [B, D, N] f32 (channel-first) -> xt_bf16 [B*N, D] bf16 (row-major)
// ---------------------------------------------------------------------------
__global__ void transpose_x_kernel(const float* __restrict__ x,
                                   __bf16* __restrict__ xt) {
  int bd = blockIdx.x;              // 0 .. B*D-1
  int b = bd / Dd;
  const float* src = x + (size_t)bd * Nn;
  for (int n = threadIdx.x; n < Nn; n += blockDim.x) {
    xt[((size_t)(b * Nn + n)) * Dd + (bd % Dd)] = (__bf16)src[n];
  }
}

// ---------------------------------------------------------------------------
// implicit = codebook @ W^T   (C x D) = (C x D) x (D x D)^T
// one wave per 16x16 output tile, bf16 WMMA, f32 accum.
// ---------------------------------------------------------------------------
__global__ void __launch_bounds__(256)
codebook_transform_kernel(const __bf16* __restrict__ cb,
                          const __bf16* __restrict__ w,
                          float* __restrict__ impl_f32,
                          __bf16* __restrict__ impl_bf) {
  int wave = threadIdx.x >> 5;
  int lane = threadIdx.x & 31;
  int tile = blockIdx.x * 8 + wave;
  int ctile = tile >> 5;            // 512 tiles along C (8192/16)
  int dtile = tile & 31;            // 32 tiles along D (512/16)
  int hl  = lane >> 4;              // K-half select
  int l15 = lane & 15;

  const __bf16* ap = cb + (size_t)(ctile * 16 + l15) * Dd;  // A row
  const __bf16* bp = w  + (size_t)(dtile * 16 + l15) * Dd;  // B col = W row (W^T)
  int aoff = hl * 16;
  int boff = hl * 16;

  v8f acc = {};
  #pragma unroll
  for (int s = 0; s < Dd / 32; ++s) {
    v8bf a0 = *(const v8bf*)(ap + s * 32 + aoff);
    v8bf a1 = *(const v8bf*)(ap + s * 32 + aoff + 8);
    v16bf a;
    #pragma unroll
    for (int i = 0; i < 8; ++i) { a[i] = a0[i]; a[i + 8] = a1[i]; }
    v16bf bfrag = *(const v16bf*)(bp + s * 32 + boff);
    acc = __builtin_amdgcn_wmma_f32_16x16x32_bf16(
        false, a, false, bfrag, (short)0, acc, false, false);
  }

  #pragma unroll
  for (int v = 0; v < 8; ++v) {
    int m = ctile * 16 + v + hl * 8;      // C/D layout: M = v (+8 for upper half)
    int n = dtile * 16 + l15;             // N = lane&15
    float val = acc[v];
    impl_f32[(size_t)m * Dd + n] = val;
    impl_bf [(size_t)m * Dd + n] = (__bf16)val;
  }
}

// ---------------------------------------------------------------------------
// half_csq[c] = 0.5 * ||implicit[c]||^2   (one wave per row)
// ---------------------------------------------------------------------------
__global__ void half_csq_kernel(const float* __restrict__ impl,
                                float* __restrict__ half_csq) {
  int wave = threadIdx.x >> 5, lane = threadIdx.x & 31;
  int row = blockIdx.x * 8 + wave;
  const float* p = impl + (size_t)row * Dd;
  float s = 0.f;
  for (int i = lane; i < Dd; i += 32) { float v = p[i]; s += v * v; }
  #pragma unroll
  for (int off = 16; off >= 1; off >>= 1) s += __shfl_xor(s, off, 32);
  if (lane == 0) half_csq[row] = 0.5f * s;
}

// ---------------------------------------------------------------------------
// TDM helper: issue tensor_load_to_lds of a [16 x 512] bf16 codebook tile
// (row-major, dim0 = 512 elems contiguous) into LDS at lds_byte_off.
// D# layout per CDNA5 ISA ch.8 (2-D tensor: groups 2/3 zero).
// clang-23 toolchain: 6-arg builtin (g0, g1, g2, g3, g4, cpol).
// ---------------------------------------------------------------------------
__device__ __forceinline__ void tdm_load_tile(const __bf16* gptr,
                                              unsigned lds_byte_off) {
  unsigned long long ga = (unsigned long long)(uintptr_t)gptr;
  u32x4 g0;
  g0[0] = 1u;                                        // count=1 valid user D#
  g0[1] = lds_byte_off;                              // lds_addr
  g0[2] = (unsigned)(ga & 0xFFFFFFFFu);              // global_addr[31:0]
  g0[3] = (unsigned)((ga >> 32) & 0x01FFFFFFu)       // global_addr[56:32]
        | (2u << 30);                                // type = 2 (image)
  i32x8 g1;
  g1[0] = (1 << 16);                                 // data_size=1 -> 2 bytes
  g1[1] = (int)((unsigned)(Dd & 0xFFFF) << 16);      // tensor_dim0[15:0]
  g1[2] = (int)(((unsigned)Dd >> 16) |               // tensor_dim0[31:16]
                ((unsigned)(Cc & 0xFFFF) << 16));    // tensor_dim1[15:0]
  g1[3] = (int)(((unsigned)Cc >> 16) |               // tensor_dim1[31:16]
                ((unsigned)Dd << 16));               // tile_dim0 = 512
  g1[4] = 16;                                        // tile_dim1=16, tile_dim2=0
  g1[5] = Dd;                                        // tensor_dim0_stride lo
  g1[6] = 0;                                         // stride hi / dim1_stride lo
  g1[7] = 0;
  i32x4 gz4 = {0, 0, 0, 0};
  i32x8 gz8 = {0, 0, 0, 0, 0, 0, 0, 0};
  __builtin_amdgcn_tensor_load_to_lds(g0, g1, gz4, gz4, gz8, 0);
}

// ---------------------------------------------------------------------------
// Fused distance-GEMM + argmin:  argmin_c ||x-c||^2 == argmax_c (x.c - 0.5c^2)
// Each wave owns 16 rows; A fragment (16x512 bf16) register-resident and
// reused for all 8192 codes. Codebook tiles (16 codes x 512) are streamed by
// the Tensor Data Mover into double-buffered LDS (one TDM issue per tile,
// consumed by all 8 waves), overlapped with the 16 WMMAs of the previous tile.
// ---------------------------------------------------------------------------
__global__ void __launch_bounds__(256, 1)
nn_argmin_kernel(const __bf16* __restrict__ xt,
                 const __bf16* __restrict__ impl_bf,
                 const float* __restrict__ half_csq,
                 int* __restrict__ idx_out,
                 float* __restrict__ idx_f_out) {
  __shared__ __align__(128) __bf16 lbuf[2][16 * Dd];   // 2 x 16KB tiles

  int wave = threadIdx.x >> 5, lane = threadIdx.x & 31;
  int hl = lane >> 4, l15 = lane & 15;
  int rowbase = blockIdx.x * 128 + wave * 16;

  // LDS byte offsets of the two buffers (flat addr low 32 bits = LDS offset)
  unsigned lds0 = (unsigned)(uintptr_t)(&lbuf[0][0]);
  unsigned lds1 = (unsigned)(uintptr_t)(&lbuf[1][0]);

  // Load A fragments once: row = rowbase + (lane&15), contiguous K: lane half hl holds k 16*hl..16*hl+15 of each 32-step.
  v16bf afrag[16];
  const __bf16* ap = xt + (size_t)(rowbase + l15) * Dd + hl * 16;
  #pragma unroll
  for (int s = 0; s < 16; ++s) {
    v8bf a0 = *(const v8bf*)(ap + s * 32);
    v8bf a1 = *(const v8bf*)(ap + s * 32 + 8);
    #pragma unroll
    for (int i = 0; i < 8; ++i) { afrag[s][i] = a0[i]; afrag[s][i + 8] = a1[i]; }
  }

  float best[8]; int bidx[8];
  #pragma unroll
  for (int v = 0; v < 8; ++v) { best[v] = -3.4e38f; bidx[v] = 0; }

  const int NT = Cc / 16;                 // 512 code tiles
  // Prologue: TDM-load tile 0 into buffer 0.
  if (wave == 0) {
    tdm_load_tile(impl_bf, lds0);
    __builtin_amdgcn_s_wait_tensorcnt(0);
  }
  __syncthreads();

  for (int ct = 0; ct < NT; ++ct) {
    int cur = ct & 1;
    // Issue next tile's DMA while computing on the current one.
    if (wave == 0 && ct + 1 < NT)
      tdm_load_tile(impl_bf + (size_t)(ct + 1) * 16 * Dd, cur ? lds0 : lds1);

    const __bf16* bp = &lbuf[cur][(size_t)l15 * Dd + hl * 16];
    v8f acc = {};
    #pragma unroll
    for (int s = 0; s < 16; ++s) {
      v16bf bfrag = *(const v16bf*)(bp + s * 32);
      acc = __builtin_amdgcn_wmma_f32_16x16x32_bf16(
          false, afrag[s], false, bfrag, (short)0, acc, false, false);
    }

    int code = ct * 16 + l15;
    float hc = half_csq[code];
    #pragma unroll
    for (int v = 0; v < 8; ++v) {
      float sc = acc[v] - hc;
      if (sc > best[v]) { best[v] = sc; bidx[v] = code; }
    }

    if (wave == 0 && ct + 1 < NT)
      __builtin_amdgcn_s_wait_tensorcnt(0);     // next buffer landed
    __syncthreads();                            // publish next / protect reuse
  }

  // Argmax reduce across the 16 lanes of each half (wave32 shuffles).
  #pragma unroll
  for (int off = 8; off >= 1; off >>= 1) {
    #pragma unroll
    for (int v = 0; v < 8; ++v) {
      float ob = __shfl_xor(best[v], off, 32);
      int   oi = __shfl_xor(bidx[v], off, 32);
      if (ob > best[v] || (ob == best[v] && oi < bidx[v])) {
        best[v] = ob; bidx[v] = oi;
      }
    }
  }
  if (l15 == 0) {
    #pragma unroll
    for (int v = 0; v < 8; ++v) {
      int m = rowbase + v + hl * 8;
      idx_out[m]   = bidx[v];
      idx_f_out[m] = (float)bidx[v];
    }
  }
}

// ---------------------------------------------------------------------------
// quantized_cf[b,d,n] = implicit[idx[b,n], d]; accumulate commit-loss sum.
// ---------------------------------------------------------------------------
__global__ void gather_loss_kernel(const float* __restrict__ x,
                                   const float* __restrict__ impl,
                                   const int* __restrict__ idx,
                                   float* __restrict__ qout,
                                   float* __restrict__ loss_accum) {
  int m = blockIdx.x;               // 0 .. 16383
  int b = m / Nn, n = m % Nn;
  const float* crow = impl + (size_t)idx[m] * Dd;
  const float* xrow = x    + (size_t)b * Dd * Nn + n;
  float*       qrow = qout + (size_t)b * Dd * Nn + n;
  float lsum = 0.f;
  for (int d = threadIdx.x; d < Dd; d += blockDim.x) {
    float q = crow[d];
    qrow[(size_t)d * Nn] = q;
    float diff = xrow[(size_t)d * Nn] - q;
    lsum += diff * diff;
  }
  #pragma unroll
  for (int off = 16; off >= 1; off >>= 1) lsum += __shfl_xor(lsum, off, 32);
  __shared__ float red[8];
  if ((threadIdx.x & 31) == 0) red[threadIdx.x >> 5] = lsum;
  __syncthreads();
  if (threadIdx.x == 0) {
    float t = 0.f;
    #pragma unroll
    for (int i = 0; i < 8; ++i) t += red[i];
    atomicAdd(loss_accum, t);
  }
}

__global__ void init_loss_kernel(float* loss_accum) { *loss_accum = 0.f; }

__global__ void finalize_loss_kernel(const float* __restrict__ loss_accum,
                                     float* __restrict__ out) {
  out[0] = (*loss_accum) * (1.25f / (float)((size_t)Bn * Nn * Dd));
}

// ---------------------------------------------------------------------------
extern "C" void kernel_launch(void* const* d_in, const int* in_sizes, int n_in,
                              void* d_out, int out_size, void* d_ws, size_t ws_size,
                              hipStream_t stream) {
  const float* x  = (const float*)d_in[0];   // [B, D, N]
  const float* cb = (const float*)d_in[1];   // [C, D]
  const float* W  = (const float*)d_in[2];   // [D, D]
  float* out = (float*)d_out;                // [quantized_cf | indices | loss]

  char* ws = (char*)d_ws;
  size_t off = 0;
  auto alloc = [&](size_t bytes) -> void* {
    void* p = ws + off;
    off += (bytes + 255) & ~(size_t)255;
    return p;
  };
  __bf16* cb_bf   = (__bf16*)alloc((size_t)Cc * Dd * 2);
  __bf16* w_bf    = (__bf16*)alloc((size_t)Dd * Dd * 2);
  float*  impl_f  = (float*) alloc((size_t)Cc * Dd * 4);
  __bf16* impl_bf = (__bf16*)alloc((size_t)Cc * Dd * 2);
  float*  hcsq    = (float*) alloc((size_t)Cc * 4);
  __bf16* xt_bf   = (__bf16*)alloc((size_t)Mrows * Dd * 2);
  int*    idxws   = (int*)   alloc((size_t)Mrows * 4);
  float*  lacc    = (float*) alloc(256);

  f32_to_bf16_kernel<<<(Cc * Dd + 255) / 256, 256, 0, stream>>>(cb, cb_bf, Cc * Dd);
  f32_to_bf16_kernel<<<(Dd * Dd + 255) / 256, 256, 0, stream>>>(W, w_bf, Dd * Dd);
  transpose_x_kernel<<<Bn * Dd, 256, 0, stream>>>(x, xt_bf);
  codebook_transform_kernel<<<(Cc / 16) * (Dd / 16) / 8, 256, 0, stream>>>(
      cb_bf, w_bf, impl_f, impl_bf);
  half_csq_kernel<<<Cc / 8, 256, 0, stream>>>(impl_f, hcsq);
  init_loss_kernel<<<1, 1, 0, stream>>>(lacc);
  nn_argmin_kernel<<<Mrows / 128, 256, 0, stream>>>(xt_bf, impl_bf, hcsq,
                                                    idxws, out + QSIZE);
  gather_loss_kernel<<<Mrows, 256, 0, stream>>>(x, impl_f, idxws, out, lacc);
  finalize_loss_kernel<<<1, 1, 0, stream>>>(lacc, out + QSIZE + Mrows);
}
